// Attention_49598282334398
// MI455X (gfx1250) — matrix operand order repaired and checked by reading, not compiled, not run on hardware
//
#include <hip/hip_runtime.h>

#define NDIM 1024
#define HEADS 16
#define DHEAD 64
#define SEQ 2048
#define BATCH 2
#define TOKENS (BATCH*SEQ)     /* 4096 */
#define NQKV (3*HEADS*DHEAD)   /* 3072 */
#define ATTN_SCALE 0.125f      /* 64^-0.5 */

typedef __attribute__((ext_vector_type(16))) __bf16 v16bf;
typedef __attribute__((ext_vector_type(8)))  float  v8f;

union FragBF { v16bf v; unsigned int u[8]; unsigned short s[16]; };

__device__ inline unsigned short f32_to_bf16(float f) {
    unsigned int u = __float_as_uint(f);
    u += 0x7FFFu + ((u >> 16) & 1u);           // round to nearest even
    return (unsigned short)(u >> 16);
}

__device__ inline v8f vzero8() {
    v8f z;
#pragma unroll
    for (int i = 0; i < 8; ++i) z[i] = 0.0f;
    return z;
}

__device__ inline v8f wmma_bf16(const FragBF& a, const FragBF& b, v8f c) {
    return __builtin_amdgcn_wmma_f32_16x16x32_bf16(
        false, a.v, false, b.v, (short)0, c, false, false);
}

// ---------------------------------------------------------------- f32 -> bf16
__global__ void cvt_bf16_kernel(const float* __restrict__ in,
                                unsigned short* __restrict__ out, long n) {
    long i = (long)blockIdx.x * blockDim.x + threadIdx.x;
    if (i < n) out[i] = f32_to_bf16(in[i]);
}

// ------------------------------------------------- RMSNorm + mix/gates (fused)
// one block per token (256 threads, 1024 dims)
__global__ __launch_bounds__(256) void norm_mix_kernel(
    const float* __restrict__ x, const float* __restrict__ gamma,
    const float* __restrict__ w_mix, const float* __restrict__ b_mix,
    const float* __restrict__ w_gates, const float* __restrict__ b_gates,
    unsigned short* __restrict__ xn_bf,
    float* __restrict__ mix_out, float* __restrict__ gates_out) {
    __shared__ float red[256];
    __shared__ float accM[HEADS];
    __shared__ float accG[HEADS];
    int t = threadIdx.x;
    long tok = blockIdx.x;
    const float* xr = x + tok * NDIM;

    float xv[4];
    float ss = 0.0f;
#pragma unroll
    for (int e = 0; e < 4; ++e) {
        xv[e] = xr[t + e * 256];
        ss += xv[e] * xv[e];
    }
    red[t] = ss;
    if (t < HEADS) { accM[t] = 0.0f; accG[t] = 0.0f; }
    __syncthreads();
    for (int off = 128; off > 0; off >>= 1) {
        if (t < off) red[t] += red[t + off];
        __syncthreads();
    }
    float norm = sqrtf(red[0]);
    float rs = 32.0f / fmaxf(norm, 1e-12f);   // sqrt(1024)=32

    float pm[HEADS], pg[HEADS];
#pragma unroll
    for (int h = 0; h < HEADS; ++h) { pm[h] = 0.0f; pg[h] = 0.0f; }

#pragma unroll
    for (int e = 0; e < 4; ++e) {
        int d = t + e * 256;
        float xn = xv[e] * rs * gamma[d];
        xn_bf[tok * NDIM + d] = f32_to_bf16(xn);
#pragma unroll
        for (int h = 0; h < HEADS; ++h) {
            pm[h] += xn * w_mix[d * HEADS + h];
            pg[h] += xn * w_gates[d * HEADS + h];
        }
    }
#pragma unroll
    for (int h = 0; h < HEADS; ++h) {
        atomicAdd(&accM[h], pm[h]);
        atomicAdd(&accG[h], pg[h]);
    }
    __syncthreads();
    if (t < HEADS) {
        int nn = (int)(tok & (SEQ - 1));
        int b = (int)(tok >> 11);
        long idx = (((long)b * HEADS + t) * SEQ) + nn;
        mix_out[idx]   = 1.0f / (1.0f + __expf(-(accM[t] + b_mix[t])));
        gates_out[idx] = 1.0f / (1.0f + __expf(-(accG[t] + b_gates[t])));
    }
}

// ----------------------------------------------------------- QKV GEMM (WMMA)
// A: xn_bf [4096][1024], B: wqkv_bf [1024][3072]; scatter epilogue.
// B tile staged in LDS pre-packed per column: Bt[col][j] = (k=2j, k=2j+1),
// so a B fragment is 8 contiguous uints -> ds_load_b128 pairs.
__global__ __launch_bounds__(256) void qkv_gemm_kernel(
    const unsigned short* __restrict__ A, const unsigned short* __restrict__ B,
    float* __restrict__ q_out, float* __restrict__ k_out,
    float* __restrict__ v_out /* = d_out part 2 */) {
    __shared__ __align__(16) unsigned int As[128 * 16];     // 128 rows x 32 bf16
    __shared__ __align__(16) unsigned int Bt[128 * 16];     // 128 cols x 32 k (packed)

    int tid = threadIdx.x;
    int lane = tid & 31, wave = tid >> 5;
    int wm = wave >> 1, wn = wave & 1;
    int lcol = lane & 15, khalf = lane >> 4;
    int bm = blockIdx.x * 128;
    int bn = blockIdx.y * 128;

    v8f acc[2][4];
#pragma unroll
    for (int mt = 0; mt < 2; ++mt)
#pragma unroll
        for (int nt = 0; nt < 4; ++nt) acc[mt][nt] = vzero8();

    for (int k0 = 0; k0 < NDIM; k0 += 32) {
        __syncthreads();
        {
            const uint4* Ag = (const uint4*)A;          // row stride 128 uint4
#pragma unroll
            for (int q = 0; q < 2; ++q) {
                int li = tid + q * 256;
                int r = li >> 2, c = li & 3;
                ((uint4*)As)[li] = Ag[(long)(bm + r) * 128 + (k0 >> 3) + c];
            }
            // B: pack 2x2 blocks (rows k,k+1 x cols 2c,2c+1) into frag order
#pragma unroll
            for (int q = 0; q < 4; ++q) {
                int li = tid + q * 256;     // 1024 cells
                int kp = li >> 6;           // k pair 0..15
                int cp = li & 63;           // col pair 0..63
                unsigned int a0 = *(const unsigned int*)
                    (B + (long)(k0 + 2 * kp) * NQKV + bn + 2 * cp);
                unsigned int a1 = *(const unsigned int*)
                    (B + (long)(k0 + 2 * kp + 1) * NQKV + bn + 2 * cp);
                Bt[(2 * cp) * 16 + kp]     = (a0 & 0xFFFFu) | (a1 << 16);
                Bt[(2 * cp + 1) * 16 + kp] = (a0 >> 16) | (a1 & 0xFFFF0000u);
            }
        }
        __syncthreads();

        FragBF a[2];
#pragma unroll
        for (int mt = 0; mt < 2; ++mt) {
            const unsigned int* arow = &As[(wm * 32 + mt * 16 + lcol) * 16];
#pragma unroll
            for (int p = 0; p < 8; ++p)
                a[mt].u[p] = arow[khalf * 8 + p];
        }
#pragma unroll
        for (int nt = 0; nt < 4; ++nt) {
            FragBF bf;
            const unsigned int* bcol = &Bt[(wn * 64 + nt * 16 + lcol) * 16 + khalf * 8];
#pragma unroll
            for (int p = 0; p < 8; ++p) bf.u[p] = bcol[p];
#pragma unroll
            for (int mt = 0; mt < 2; ++mt)
                acc[mt][nt] = wmma_bf16(a[mt], bf, acc[mt][nt]);
        }
    }

#pragma unroll
    for (int mt = 0; mt < 2; ++mt)
#pragma unroll
        for (int nt = 0; nt < 4; ++nt) {
            int col = bn + wn * 64 + nt * 16 + lcol;
            int i3 = col >> 10;
            int h = (col >> 6) & 15;
            int dd = col & 63;
#pragma unroll
            for (int r = 0; r < 8; ++r) {
                int row = bm + wm * 32 + mt * 16 + r + 8 * khalf;
                int bb = row >> 11, nn = row & (SEQ - 1);
                long idx = ((((long)bb * HEADS + h) * SEQ) + nn) * DHEAD + dd;
                float val = acc[mt][nt][r];
                if (i3 == 0) q_out[idx] = val;
                else if (i3 == 1) k_out[idx] = val;
                else v_out[idx] = val;
            }
        }
}

// ------------------------------------------- RoPE + value-residual mix -> bf16
__global__ __launch_bounds__(256) void rope_mix_kernel(
    const float* __restrict__ qf, const float* __restrict__ kf,
    const float* __restrict__ v_orig, const float* __restrict__ vres,
    const float* __restrict__ mix,
    unsigned short* __restrict__ qbf, unsigned short* __restrict__ kbf,
    unsigned short* __restrict__ vbf) {
    long id = (long)blockIdx.x * blockDim.x + threadIdx.x;
    const long total = (long)BATCH * HEADS * SEQ * (DHEAD / 2);
    if (id >= total) return;
    int i = (int)(id & 31);            // freq pair index
    long th = id >> 5;                 // (b*16+h)*2048+nn
    int nn = (int)(th & (SEQ - 1));

    float expo = (float)(2 * i) * (1.0f / 64.0f);
    float invf = __powf(10000.0f, -expo);
    float ang = (float)nn * invf;
    float s_, c;
    __sincosf(ang, &s_, &c);

    long base = th * DHEAD + 2 * i;
    float q1 = qf[base], q2 = qf[base + 1];
    qbf[base]     = f32_to_bf16(q1 * c - q2 * s_);
    qbf[base + 1] = f32_to_bf16(q1 * s_ + q2 * c);
    float k1 = kf[base], k2 = kf[base + 1];
    kbf[base]     = f32_to_bf16(k1 * c - k2 * s_);
    kbf[base + 1] = f32_to_bf16(k1 * s_ + k2 * c);

    float mx = mix[th];
    float v1 = v_orig[base], v2 = v_orig[base + 1];
    vbf[base]     = f32_to_bf16(v1 * (1.0f - mx) + vres[base] * mx);
    vbf[base + 1] = f32_to_bf16(v2 * (1.0f - mx) + vres[base + 1] * mx);
}

// ------------------------------------------------ flash attention (WMMA, bf16)
// grid (32 qblocks, 32 b*h); 128 threads = 4 waves; each wave owns 16 q rows.
__global__ __launch_bounds__(128) void attn_kernel(
    const unsigned short* __restrict__ qbf, const unsigned short* __restrict__ kbf,
    const unsigned short* __restrict__ vbf, const float* __restrict__ gates,
    unsigned short* __restrict__ attn_bf) {
    __shared__ __align__(16) unsigned short Ks[64 * 64];      // [j][d]
    __shared__ __align__(16) unsigned short Vt[64 * 64];      // [d][j]
    __shared__ __align__(16) unsigned short Ps[4][16 * 64];   // per-wave P tile

    int tid = threadIdx.x, lane = tid & 31, wave = tid >> 5;
    int lcol = lane & 15, khalf = lane >> 4;
    int qb = blockIdx.x;
    int bh = blockIdx.y;
    const unsigned short* qB = qbf + (long)bh * SEQ * DHEAD;
    const unsigned short* kB = kbf + (long)bh * SEQ * DHEAD;
    const unsigned short* vB = vbf + (long)bh * SEQ * DHEAD;
    int qrow0 = qb * 64 + wave * 16;

    FragBF qfrag[2];
    {
        const unsigned int* qrow = (const unsigned int*)(qB + (long)(qrow0 + lcol) * DHEAD);
#pragma unroll
        for (int dh = 0; dh < 2; ++dh)
#pragma unroll
            for (int p = 0; p < 8; ++p)
                qfrag[dh].u[p] = qrow[dh * 16 + khalf * 8 + p];
    }

    v8f acc[4];
#pragma unroll
    for (int dt = 0; dt < 4; ++dt) acc[dt] = vzero8();
    float m_[8], l_[8];
#pragma unroll
    for (int r = 0; r < 8; ++r) { m_[r] = -3.0e38f; l_[r] = 0.0f; }

    for (int kb = 0; kb < SEQ / 64; ++kb) {
        __syncthreads();
        {   // load K tile (row-major) and V tile (transposed)
            const uint4* kg = (const uint4*)(kB + (long)kb * 64 * DHEAD);
#pragma unroll
            for (int q = 0; q < 4; ++q) {
                int li = tid + q * 128;
                ((uint4*)Ks)[li] = kg[li];
            }
            const unsigned int* vg = (const unsigned int*)(vB + (long)kb * 64 * DHEAD);
#pragma unroll
            for (int q = 0; q < 16; ++q) {
                int li = tid + q * 128;          // [0, 2048)
                int r = li >> 5;                 // v row j
                int c = li & 31;                 // d pair
                unsigned int d2 = vg[li];
                Vt[(2 * c) * 64 + r]     = (unsigned short)(d2 & 0xFFFFu);
                Vt[(2 * c + 1) * 64 + r] = (unsigned short)(d2 >> 16);
            }
        }
        __syncthreads();

        // S = Q K^T
        v8f s[4];
#pragma unroll
        for (int nt = 0; nt < 4; ++nt) {
            s[nt] = vzero8();
#pragma unroll
            for (int dh = 0; dh < 2; ++dh) {
                FragBF kf;
                const unsigned int* krow = (const unsigned int*)&Ks[(nt * 16 + lcol) * 64];
#pragma unroll
                for (int p = 0; p < 8; ++p)
                    kf.u[p] = krow[dh * 16 + khalf * 8 + p];
                s[nt] = wmma_bf16(qfrag[dh], kf, s[nt]);
            }
        }

        // online softmax
        float alpha[8];
#pragma unroll
        for (int r = 0; r < 8; ++r) {
            float mx = -3.0e38f;
#pragma unroll
            for (int nt = 0; nt < 4; ++nt) {
                s[nt][r] *= ATTN_SCALE;
                mx = fmaxf(mx, s[nt][r]);
            }
            for (int off = 8; off >= 1; off >>= 1)
                mx = fmaxf(mx, __shfl_xor(mx, off, 32));
            float mnew = fmaxf(m_[r], mx);
            float a = __expf(m_[r] - mnew);
            float rs = 0.0f;
#pragma unroll
            for (int nt = 0; nt < 4; ++nt) {
                float p = __expf(s[nt][r] - mnew);
                s[nt][r] = p;
                rs += p;
            }
            for (int off = 8; off >= 1; off >>= 1)
                rs += __shfl_xor(rs, off, 32);
            l_[r] = l_[r] * a + rs;
            m_[r] = mnew;
            alpha[r] = a;
        }
#pragma unroll
        for (int dt = 0; dt < 4; ++dt)
#pragma unroll
            for (int r = 0; r < 8; ++r) acc[dt][r] *= alpha[r];

        // P tile: C layout -> LDS -> A layout
        unsigned short* pw = &Ps[wave][0];
#pragma unroll
        for (int nt = 0; nt < 4; ++nt)
#pragma unroll
            for (int r = 0; r < 8; ++r)
                pw[(r + 8 * khalf) * 64 + nt * 16 + lcol] = f32_to_bf16(s[nt][r]);
        __syncthreads();

        // O += P V
#pragma unroll
        for (int dt = 0; dt < 4; ++dt) {
#pragma unroll
            for (int jh = 0; jh < 2; ++jh) {
                FragBF pf, vf;
                const unsigned int* prow = (const unsigned int*)&pw[lcol * 64];
#pragma unroll
                for (int p = 0; p < 8; ++p)
                    pf.u[p] = prow[jh * 16 + khalf * 8 + p];
                const unsigned int* vrow = (const unsigned int*)&Vt[(dt * 16 + lcol) * 64];
#pragma unroll
                for (int p = 0; p < 8; ++p)
                    vf.u[p] = vrow[jh * 16 + khalf * 8 + p];
                acc[dt] = wmma_bf16(pf, vf, acc[dt]);
            }
        }
    }

    int b = bh >> 4, h = bh & 15;
#pragma unroll
    for (int r = 0; r < 8; ++r) {
        int row = qrow0 + r + 8 * khalf;
        float inv = 1.0f / l_[r];
        float g = gates[(long)bh * SEQ + row];
#pragma unroll
        for (int dt = 0; dt < 4; ++dt) {
            float o = acc[dt][r] * inv * g;
            attn_bf[(((long)b * SEQ + row) * NDIM) + h * DHEAD + dt * 16 + lcol] =
                f32_to_bf16(o);
        }
    }
}

// ----------------------------------------------------- output GEMM (WMMA)
// A: attn_bf [4096][1024], B: wout_bf [1024][1024] -> d_out f32 [4096][1024]
__global__ __launch_bounds__(256) void out_gemm_kernel(
    const unsigned short* __restrict__ A, const unsigned short* __restrict__ B,
    float* __restrict__ C) {
    __shared__ __align__(16) unsigned int As[128 * 16];
    __shared__ __align__(16) unsigned int Bt[128 * 16];

    int tid = threadIdx.x;
    int lane = tid & 31, wave = tid >> 5;
    int wm = wave >> 1, wn = wave & 1;
    int lcol = lane & 15, khalf = lane >> 4;
    int bm = blockIdx.x * 128;
    int bn = blockIdx.y * 128;

    v8f acc[2][4];
#pragma unroll
    for (int mt = 0; mt < 2; ++mt)
#pragma unroll
        for (int nt = 0; nt < 4; ++nt) acc[mt][nt] = vzero8();

    for (int k0 = 0; k0 < NDIM; k0 += 32) {
        __syncthreads();
        {
            const uint4* Ag = (const uint4*)A;          // row stride 128 uint4
#pragma unroll
            for (int q = 0; q < 2; ++q) {
                int li = tid + q * 256;
                int r = li >> 2, c = li & 3;
                ((uint4*)As)[li] = Ag[(long)(bm + r) * 128 + (k0 >> 3) + c];
            }
#pragma unroll
            for (int q = 0; q < 4; ++q) {
                int li = tid + q * 256;
                int kp = li >> 6;
                int cp = li & 63;
                unsigned int a0 = *(const unsigned int*)
                    (B + (long)(k0 + 2 * kp) * NDIM + bn + 2 * cp);
                unsigned int a1 = *(const unsigned int*)
                    (B + (long)(k0 + 2 * kp + 1) * NDIM + bn + 2 * cp);
                Bt[(2 * cp) * 16 + kp]     = (a0 & 0xFFFFu) | (a1 << 16);
                Bt[(2 * cp + 1) * 16 + kp] = (a0 >> 16) | (a1 & 0xFFFF0000u);
            }
        }
        __syncthreads();

        FragBF a[2];
#pragma unroll
        for (int mt = 0; mt < 2; ++mt) {
            const unsigned int* arow = &As[(wm * 32 + mt * 16 + lcol) * 16];
#pragma unroll
            for (int p = 0; p < 8; ++p)
                a[mt].u[p] = arow[khalf * 8 + p];
        }
#pragma unroll
        for (int nt = 0; nt < 4; ++nt) {
            FragBF bf;
            const unsigned int* bcol = &Bt[(wn * 64 + nt * 16 + lcol) * 16 + khalf * 8];
#pragma unroll
            for (int p = 0; p < 8; ++p) bf.u[p] = bcol[p];
#pragma unroll
            for (int mt = 0; mt < 2; ++mt)
                acc[mt][nt] = wmma_bf16(a[mt], bf, acc[mt][nt]);
        }
    }

#pragma unroll
    for (int mt = 0; mt < 2; ++mt)
#pragma unroll
        for (int nt = 0; nt < 4; ++nt) {
            int col = bn + wn * 64 + nt * 16 + lcol;
#pragma unroll
            for (int r = 0; r < 8; ++r) {
                int row = bm + wm * 32 + mt * 16 + r + 8 * khalf;
                C[(long)row * NDIM + col] = acc[mt][nt][r];
            }
        }
}

// ============================================================== host launcher
extern "C" void kernel_launch(void* const* d_in, const int* in_sizes, int n_in,
                              void* d_out, int out_size, void* d_ws, size_t ws_size,
                              hipStream_t stream) {
    const float* x       = (const float*)d_in[0];
    const float* vres    = (const float*)d_in[1];
    const float* gamma   = (const float*)d_in[2];
    const float* w_qkv   = (const float*)d_in[3];
    const float* w_mix   = (const float*)d_in[4];
    const float* b_mix   = (const float*)d_in[5];
    const float* w_gates = (const float*)d_in[6];
    const float* b_gates = (const float*)d_in[7];
    const float* w_out   = (const float*)d_in[8];

    float* out1  = (float*)d_out;                        // [b][n][1024]
    float* out_v = out1 + (size_t)TOKENS * NDIM;         // [b][h][n][64]

    char* ws = (char*)d_ws;
    size_t off = 0;
    auto wsalloc = [&](size_t bytes) {
        void* p = ws + off;
        off += (bytes + 255) & ~(size_t)255;
        return p;
    };
    unsigned short* wqkv_bf = (unsigned short*)wsalloc((size_t)NDIM * NQKV * 2);
    unsigned short* wout_bf = (unsigned short*)wsalloc((size_t)NDIM * NDIM * 2);
    unsigned short* xn_bf   = (unsigned short*)wsalloc((size_t)TOKENS * NDIM * 2);
    float* q_f  = (float*)wsalloc((size_t)TOKENS * NDIM * 4);
    float* k_f  = (float*)wsalloc((size_t)TOKENS * NDIM * 4);
    unsigned short* q_bf = (unsigned short*)wsalloc((size_t)TOKENS * NDIM * 2);
    unsigned short* k_bf = (unsigned short*)wsalloc((size_t)TOKENS * NDIM * 2);
    unsigned short* v_bf = (unsigned short*)wsalloc((size_t)TOKENS * NDIM * 2);
    float* mix_ws   = (float*)wsalloc((size_t)BATCH * HEADS * SEQ * 4);
    float* gates_ws = (float*)wsalloc((size_t)BATCH * HEADS * SEQ * 4);
    unsigned short* attn_bf = (unsigned short*)wsalloc((size_t)TOKENS * NDIM * 2);

    // 1) weight conversion
    {
        long n1 = (long)NDIM * NQKV;
        cvt_bf16_kernel<<<(unsigned)((n1 + 255) / 256), 256, 0, stream>>>(w_qkv, wqkv_bf, n1);
        long n2 = (long)NDIM * NDIM;
        cvt_bf16_kernel<<<(unsigned)((n2 + 255) / 256), 256, 0, stream>>>(w_out, wout_bf, n2);
    }
    // 2) rmsnorm + mix/gates
    norm_mix_kernel<<<TOKENS, 256, 0, stream>>>(x, gamma, w_mix, b_mix, w_gates, b_gates,
                                                xn_bf, mix_ws, gates_ws);
    // 3) QKV GEMM (v -> d_out part 2 = original_values)
    qkv_gemm_kernel<<<dim3(TOKENS / 128, NQKV / 128), 256, 0, stream>>>(
        xn_bf, wqkv_bf, q_f, k_f, out_v);
    // 4) RoPE + value residual mix
    {
        long total = (long)BATCH * HEADS * SEQ * (DHEAD / 2);
        rope_mix_kernel<<<(unsigned)((total + 255) / 256), 256, 0, stream>>>(
            q_f, k_f, out_v, vres, mix_ws, q_bf, k_bf, v_bf);
    }
    // 5) flash attention + gating
    attn_kernel<<<dim3(SEQ / 64, BATCH * HEADS), 128, 0, stream>>>(
        q_bf, k_bf, v_bf, gates_ws, attn_bf);
    // 6) output projection
    out_gemm_kernel<<<dim3(TOKENS / 128, NDIM / 128), 256, 0, stream>>>(
        attn_bf, wout_bf, out1);
}
